// PyLSTM_15496242004026
// MI455X (gfx1250) — matrix operand order repaired and checked by reading, not compiled, not run on hardware
//
#include <hip/hip_runtime.h>

#define B_    32
#define LSEQ  2048
#define DIN   256
#define HDIM  256
#define NGATE 1024   // 4*H
#define NLAY  2
#define NWG   16
#define TPB   256    // 8 waves of 32

typedef __attribute__((ext_vector_type(16))) __bf16 v16bf;
typedef __attribute__((ext_vector_type(8)))  float  v8f;

// fp32 -> bf16 round-to-nearest-even
__device__ __forceinline__ unsigned short f2bf(float f) {
  unsigned int u = __builtin_bit_cast(unsigned int, f);
  u += 0x7fffu + ((u >> 16) & 1u);
  return (unsigned short)(u >> 16);
}

union OpU { uint4 q[2]; v16bf v; };

// A operand (16x32 bf16): caller passes kb = kc*32 + half*8; dwords 0-3 at kb, 4-7 at kb+16.
__device__ __forceinline__ v16bf ld_a(const unsigned short* row, int kb) {
  OpU u;
  u.q[0] = *(const uint4*)(row + kb);
  u.q[1] = *(const uint4*)(row + kb + 16);
  return u.v;
}
// B operand (32x16 bf16): caller passes kb = kc*32 + half*8; dwords 0-3 at kb, 4-7 at kb+16.
__device__ __forceinline__ v16bf ld_b(const unsigned short* row, int kb) {
  OpU u;
  u.q[0] = *(const uint4*)(row + kb);
  u.q[1] = *(const uint4*)(row + kb + 16);
  return u.v;
}

// CDNA5 async copy: 16 bytes global -> LDS, tracked by ASYNCcnt (no VGPR round-trip)
__device__ __forceinline__ void async_g2l_b128(void* lds_ptr, const void* gptr) {
  unsigned int lds_off = (unsigned int)(unsigned long long)lds_ptr; // low 32b of generic = LDS addr
  unsigned long long ga = (unsigned long long)gptr;
  asm volatile("global_load_async_to_lds_b128 %0, %1, off"
               :: "v"(lds_off), "v"(ga) : "memory");
}
__device__ __forceinline__ void wait_async0() {
  asm volatile("s_wait_asynccnt 0x0" ::: "memory");
}

// ---- sense-reversing grid barrier: bar[0]=count, bar[1]=generation ----
__device__ __forceinline__ void grid_barrier(unsigned int* bar) {
  __syncthreads();
  if (threadIdx.x == 0) {
    __threadfence();
    unsigned int g = __atomic_load_n(bar + 1, __ATOMIC_RELAXED);
    unsigned int arrived = atomicAdd(bar, 1u) + 1u;
    if (arrived == gridDim.x) {
      __atomic_store_n(bar, 0u, __ATOMIC_RELAXED);
      __atomic_store_n(bar + 1, g + 1u, __ATOMIC_RELEASE);
    } else {
      while (__atomic_load_n(bar + 1, __ATOMIC_ACQUIRE) == g) {
        __builtin_amdgcn_s_sleep(2);
      }
    }
    __threadfence();
  }
  __syncthreads();
}

// ---- init: transpose+convert weights to bf16 [l][n][k], zero h-state, reset barrier ----
__global__ void lstm_init(const float* __restrict__ Wx, const float* __restrict__ Wh,
                          unsigned short* __restrict__ Wxt, unsigned short* __restrict__ Wht,
                          unsigned short* __restrict__ hst, unsigned int* __restrict__ bar) {
  size_t idx = (size_t)blockIdx.x * blockDim.x + threadIdx.x;
  const size_t WN = (size_t)NLAY * DIN * NGATE; // 524288 per matrix
  if (idx < WN) {
    int l   = (int)(idx / ((size_t)DIN * NGATE));
    int rem = (int)(idx % ((size_t)DIN * NGATE));
    int k   = rem / NGATE;
    int n   = rem % NGATE;
    Wxt[((size_t)l * NGATE + n) * DIN + k] = f2bf(Wx[idx]);
    Wht[((size_t)l * NGATE + n) * DIN + k] = f2bf(Wh[idx]);
  }
  if (idx < (size_t)NLAY * 2 * B_ * HDIM) hst[idx] = 0;
  if (idx < 2) bar[idx] = 0;
}

// ---- persistent recurrent kernel: 16 WGs cooperate, 1 grid barrier per timestep ----
__global__ __launch_bounds__(TPB) void lstm_persist(
    const float* __restrict__ x, const float* __restrict__ bx, const float* __restrict__ bh,
    const unsigned short* __restrict__ Wxt, const unsigned short* __restrict__ Wht,
    unsigned short* __restrict__ hst, float* __restrict__ out, unsigned int* __restrict__ bar) {

  __shared__ unsigned short xbuf[B_][DIN];     // 16 KB input (bf16)
  __shared__ unsigned short hbuf[B_][HDIM];    // 16 KB h_prev (bf16)
  __shared__ float          gbuf[B_][64];      //  8 KB this WG's gate slice [i|f|g|o]x16
  __shared__ float          cbuf[NLAY][B_][16];//  4 KB cell state (fp32, persistent)

  const int tid  = threadIdx.x;
  const int lane = tid & 31;
  const int wave = tid >> 5;
  const int mt   = wave >> 2;      // row tile: 0..1  (B=32 -> 2 tiles of 16)
  const int nt   = wave & 3;       // gate index: 0=i 1=f 2=g 3=o
  const int lm   = lane & 15;
  const int half = lane >> 4;
  const int wg   = blockIdx.x;     // owns h columns [wg*16, wg*16+16)
  const int mrow = mt * 16 + lm;
  const int ncol = nt * HDIM + wg * 16 + lm;   // gate column in [0,1024)
  const int kof  = half * 8;

  const unsigned short* wxrowL[NLAY];
  const unsigned short* whrowL[NLAY];
  float biasL[NLAY];
#pragma unroll
  for (int l = 0; l < NLAY; ++l) {
    wxrowL[l] = Wxt + ((size_t)l * NGATE + ncol) * DIN;
    whrowL[l] = Wht + ((size_t)l * NGATE + ncol) * DIN;
    biasL[l]  = bx[l * NGATE + ncol] + bh[l * NGATE + ncol];
  }

  for (int i = tid; i < NLAY * B_ * 16; i += TPB) ((float*)cbuf)[i] = 0.f;
  __syncthreads();

  float* hT = out + (size_t)B_ * LSEQ * HDIM;
  float* cT = hT + (size_t)NLAY * B_ * HDIM;

#pragma unroll 1
  for (int t = 0; t < LSEQ; ++t) {
    const int wp = t & 1;        // write parity this step
    const int rp = wp ^ 1;       // read parity (h from previous step)

#pragma unroll
    for (int l = 0; l < NLAY; ++l) {
      // ---- stage A matrices into LDS (bf16) ----
      if (l == 0) {
        // h_prev(layer0): pure bf16 copy -> async global->LDS (ASYNCcnt path)
        for (int i = tid * 8; i < B_ * HDIM; i += TPB * 8) {
          int b = i >> 8, d = i & 255;
          async_g2l_b128(&hbuf[b][d],
                         &hst[(((size_t)0 * 2 + rp) * B_ + b) * HDIM + d]);
        }
        // x_t: fp32 -> bf16 conversion must go through VGPRs
        for (int i = tid * 4; i < B_ * DIN; i += TPB * 4) {
          int b = i >> 8, d = i & 255;
          const float4 v = *(const float4*)(x + ((size_t)b * LSEQ + t) * DIN + d);
          uint2 pk;
          pk.x = (unsigned int)f2bf(v.x) | ((unsigned int)f2bf(v.y) << 16);
          pk.y = (unsigned int)f2bf(v.z) | ((unsigned int)f2bf(v.w) << 16);
          *(uint2*)&xbuf[b][d] = pk;
        }
      } else {
        // layer1 input = layer0 h_t; h_prev(layer1): both pure bf16 copies -> async
        for (int i = tid * 8; i < B_ * HDIM; i += TPB * 8) {
          int b = i >> 8, d = i & 255;
          async_g2l_b128(&xbuf[b][d],
                         &hst[(((size_t)0 * 2 + wp) * B_ + b) * HDIM + d]);
          async_g2l_b128(&hbuf[b][d],
                         &hst[(((size_t)1 * 2 + rp) * B_ + b) * HDIM + d]);
        }
      }
      wait_async0();
      __syncthreads();

      // ---- gates = x_t @ Wx + h_prev @ Wh  (bf16 WMMA, fp32 accum) ----
      // Software-pipelined: issue chunk kc+1's LDS loads before chunk kc's WMMAs.
      v8f acc = {0.f, 0.f, 0.f, 0.f, 0.f, 0.f, 0.f, 0.f};
      const unsigned short* xrow = &xbuf[mrow][0];
      const unsigned short* hrow = &hbuf[mrow][0];
      v16bf ax = ld_a(xrow, kof);
      v16bf ah = ld_a(hrow, kof);
#pragma unroll
      for (int kc = 0; kc < 8; ++kc) {
        v16bf axn = ax, ahn = ah;
        if (kc < 7) {
          axn = ld_a(xrow, (kc + 1) * 32 + kof);
          ahn = ld_a(hrow, (kc + 1) * 32 + kof);
        }
        v16bf b0 = ld_b(wxrowL[l], kc * 32 + half * 8);  // register-resident after LICM
        acc = __builtin_amdgcn_wmma_f32_16x16x32_bf16(false, ax, false, b0,
                                                      (short)0, acc, false, false);
        v16bf b1 = ld_b(whrowL[l], kc * 32 + half * 8);
        acc = __builtin_amdgcn_wmma_f32_16x16x32_bf16(false, ah, false, b1,
                                                      (short)0, acc, false, false);
        ax = axn; ah = ahn;
      }
#pragma unroll
      for (int r = 0; r < 8; ++r) {
        gbuf[mt * 16 + half * 8 + r][nt * 16 + lm] = acc[r] + biasL[l];
      }
      __syncthreads();

      // ---- elementwise LSTM cell on this WG's 32x16 slice ----
#pragma unroll
      for (int rep = 0; rep < 2; ++rep) {
        int idx = tid + rep * TPB;          // 0..511
        int b  = idx >> 4;
        int jl = idx & 15;
        float iv = gbuf[b][jl];
        float fv = gbuf[b][16 + jl];
        float gv = gbuf[b][32 + jl];
        float ov = gbuf[b][48 + jl];
        float cp = cbuf[l][b][jl];
        float si = 1.f / (1.f + __expf(-iv));
        float sf = 1.f / (1.f + __expf(-fv));
        float so = 1.f / (1.f + __expf(-ov));
        float cn = sf * cp + si * tanhf(gv);
        float hn = so * tanhf(cn);
        cbuf[l][b][jl] = cn;
        int col = wg * 16 + jl;
        hst[(((size_t)l * 2 + wp) * B_ + b) * HDIM + col] = f2bf(hn);
        if (l == 1) out[((size_t)b * LSEQ + t) * HDIM + col] = hn;
        if (t == LSEQ - 1) {
          hT[((size_t)l * B_ + b) * HDIM + col] = hn;
          cT[((size_t)l * B_ + b) * HDIM + col] = cn;
        }
      }

      if (l == 0) {
        grid_barrier(bar);   // publish layer-0 h_t before layer 1 consumes it
      } else {
        __syncthreads();     // protect LDS buffers before next step's staging
      }
    }
  }
}

extern "C" void kernel_launch(void* const* d_in, const int* in_sizes, int n_in,
                              void* d_out, int out_size, void* d_ws, size_t ws_size,
                              hipStream_t stream) {
  const float* x  = (const float*)d_in[0];
  const float* Wx = (const float*)d_in[1];
  const float* Wh = (const float*)d_in[2];
  const float* bx = (const float*)d_in[3];
  const float* bh = (const float*)d_in[4];
  float* out = (float*)d_out;

  char* ws = (char*)d_ws;
  const size_t WXT_OFF = 0;
  const size_t WHT_OFF = WXT_OFF + (size_t)NLAY * NGATE * DIN * 2;   // 1 MB
  const size_t HST_OFF = WHT_OFF + (size_t)NLAY * NGATE * DIN * 2;   // 2 MB
  const size_t BAR_OFF = HST_OFF + (size_t)NLAY * 2 * B_ * HDIM * 2; // 2 MB + 64 KB
  unsigned short* Wxt = (unsigned short*)(ws + WXT_OFF);
  unsigned short* Wht = (unsigned short*)(ws + WHT_OFF);
  unsigned short* hst = (unsigned short*)(ws + HST_OFF);
  unsigned int*   bar = (unsigned int*)(ws + BAR_OFF);

  lstm_init<<<2048, 256, 0, stream>>>(Wx, Wh, Wxt, Wht, hst, bar);
  lstm_persist<<<NWG, TPB, 0, stream>>>(x, bx, bh, Wxt, Wht, hst, out, bar);
}
